// EncoderBlock_26036091749043
// MI455X (gfx1250) — matrix operand order repaired and checked by reading, not compiled, not run on hardware
//
#include <hip/hip_runtime.h>

typedef __bf16 bf16_t;
typedef __attribute__((ext_vector_type(8)))  __bf16 v8bf;
typedef __attribute__((ext_vector_type(16))) __bf16 v16bf;
typedef __attribute__((ext_vector_type(8)))  float  v8f;

struct Frag { union { v16bf v; v8bf h[2]; }; };

static __device__ __forceinline__ v8f wmma_bf16(const Frag& a, const Frag& b, v8f c) {
    return __builtin_amdgcn_wmma_f32_16x16x32_bf16(false, a.v, false, b.v, (short)0, c, false, false);
}

static __device__ __forceinline__ float gelu_exact(float x) {
    return 0.5f * x * (1.0f + erff(x * 0.70710678118654752f));
}

// CDNA5 async DMA: global -> LDS, 16 bytes per lane, tracked by ASYNCcnt
static __device__ __forceinline__ void async_copy16(unsigned ldsOff, const bf16_t* g) {
    unsigned long long ga = (unsigned long long)g;
    asm volatile("global_load_async_to_lds_b128 %0, %1, off"
                 :: "v"(ldsOff), "v"(ga) : "memory");
}
static __device__ __forceinline__ void wait_async0() {
    asm volatile("s_wait_asynccnt 0" ::: "memory");
}
static __device__ __forceinline__ unsigned lds_off(const void* p) {
    return (unsigned)(unsigned long long)p;   // low 32 bits of flat LDS address = LDS byte offset
}

// ---------------------------------------------------------------------------
// Weight convert + transpose: W[K][N] f32  ->  Wt[N][K] bf16
// ---------------------------------------------------------------------------
__global__ __launch_bounds__(256)
void transpose_to_bf16(const float* __restrict__ W, bf16_t* __restrict__ Wt, int K, int N) {
    __shared__ float tile[32][33];
    int k0 = blockIdx.y * 32, n0 = blockIdx.x * 32;
#pragma unroll
    for (int r = threadIdx.y; r < 32; r += 8)
        tile[r][threadIdx.x] = W[(size_t)(k0 + r) * N + n0 + threadIdx.x];
    __syncthreads();
#pragma unroll
    for (int r = threadIdx.y; r < 32; r += 8)
        Wt[(size_t)(n0 + r) * K + k0 + threadIdx.x] = (bf16_t)tile[threadIdx.x][r];
}

// ---------------------------------------------------------------------------
// LayerNorm (row length 1024), f32 in -> bf16 out
// ---------------------------------------------------------------------------
__global__ __launch_bounds__(256)
void layernorm_bf16_kernel(const float* __restrict__ x, const float* __restrict__ g,
                           const float* __restrict__ bta, bf16_t* __restrict__ y) {
    const int E = 1024;
    int row = blockIdx.x;
    const float* xr = x + (size_t)row * E;
    int t = threadIdx.x;
    int lane = t & 31, wv = t >> 5;

    float vals[4];
    float s = 0.f, ss = 0.f;
#pragma unroll
    for (int i = 0; i < 4; i++) {
        float f = xr[t + i * 256];
        vals[i] = f; s += f; ss += f * f;
    }
#pragma unroll
    for (int off = 16; off >= 1; off >>= 1) {
        s  += __shfl_xor(s,  off, 32);
        ss += __shfl_xor(ss, off, 32);
    }
    __shared__ float sbuf[8], ssbuf[8], mv[2];
    if (lane == 0) { sbuf[wv] = s; ssbuf[wv] = ss; }
    __syncthreads();
    if (t == 0) {
        float S = 0.f, SS = 0.f;
#pragma unroll
        for (int i = 0; i < 8; i++) { S += sbuf[i]; SS += ssbuf[i]; }
        float m = S * (1.0f / E);
        float var = SS * (1.0f / E) - m * m;
        mv[0] = m; mv[1] = rsqrtf(var + 1e-5f);
    }
    __syncthreads();
    float m = mv[0], r = mv[1];
    bf16_t* yr = y + (size_t)row * E;
#pragma unroll
    for (int i = 0; i < 4; i++) {
        int c = t + i * 256;
        yr[c] = (bf16_t)((vals[i] - m) * r * g[c] + bta[c]);
    }
}

// ---------------------------------------------------------------------------
// bf16 WMMA GEMM: C[M][N] = A[M][K] @ Bt[N][K]^T + bias, with epilogues
// EPI 0: store bf16 | 1: GELU -> bf16 | 2: +resid -> f32 | 3: V-transpose scatter bf16
// Block 128x128x32, 256 threads = 8 waves (2x4), double-buffered async-LDS staging
// ---------------------------------------------------------------------------
static __device__ __forceinline__ void gemm_stage_async(
    bf16_t* __restrict__ Asmb, bf16_t* __restrict__ Bsmb,
    const bf16_t* __restrict__ A, const bf16_t* __restrict__ Bt, int K,
    int m0, int n0, int kt, int ldrow, int ldk) {
    const bf16_t* ag = A  + (size_t)(m0 + ldrow) * K + kt + ldk;
    const bf16_t* bg = Bt + (size_t)(n0 + ldrow) * K + kt + ldk;
    unsigned ao = lds_off(&Asmb[ldrow * 40 + ldk]);
    unsigned bo = lds_off(&Bsmb[ldrow * 40 + ldk]);
    async_copy16(ao,      ag);
    async_copy16(ao + 16, ag + 8);
    async_copy16(bo,      bg);
    async_copy16(bo + 16, bg + 8);
}

static __device__ __forceinline__ void gemm_compute_step(
    const bf16_t* __restrict__ Asmb, const bf16_t* __restrict__ Bsmb,
    int wm, int wn, int hl, int l15, v8f acc[4][2]) {
    Frag af[4], bfr[2];
#pragma unroll
    for (int i = 0; i < 4; i++) {
        int r = wm * 64 + i * 16 + l15;
        af[i].h[0] = *(const v8bf*)&Asmb[r * 40 + hl * 8];
        af[i].h[1] = *(const v8bf*)&Asmb[r * 40 + 16 + hl * 8];
    }
#pragma unroll
    for (int j = 0; j < 2; j++) {
        int n = wn * 32 + j * 16 + l15;
        bfr[j].h[0] = *(const v8bf*)&Bsmb[n * 40 + hl * 8];
        bfr[j].h[1] = *(const v8bf*)&Bsmb[n * 40 + 16 + hl * 8];
    }
#pragma unroll
    for (int i = 0; i < 4; i++)
#pragma unroll
        for (int j = 0; j < 2; j++)
            acc[i][j] = wmma_bf16(af[i], bfr[j], acc[i][j]);
}

template <int EPI>
__global__ __launch_bounds__(256)
void gemm_bf16_kernel(const bf16_t* __restrict__ A, const bf16_t* __restrict__ Bt,
                      const float* __restrict__ bias, const float* __restrict__ resid,
                      void* __restrict__ Cout, int M, int N, int K) {
    __shared__ bf16_t Asm0[128 * 40];
    __shared__ bf16_t Asm1[128 * 40];
    __shared__ bf16_t Bsm0[128 * 40];
    __shared__ bf16_t Bsm1[128 * 40];

    const int tid  = threadIdx.x;
    const int lane = tid & 31;
    const int wave = tid >> 5;
    const int wm = wave & 1;        // 0..1 -> 64-row stripe
    const int wn = wave >> 1;       // 0..3 -> 32-col stripe
    const int hl  = lane >> 4;      // lane half
    const int l15 = lane & 15;

    const int m0 = blockIdx.y * 128;
    const int n0 = blockIdx.x * 128;

    const int ldrow = tid >> 1;           // 0..127
    const int ldk   = (tid & 1) * 16;     // 0 / 16

    v8f acc[4][2];
    const v8f z8 = {0.f,0.f,0.f,0.f,0.f,0.f,0.f,0.f};
#pragma unroll
    for (int i = 0; i < 4; i++)
#pragma unroll
        for (int j = 0; j < 2; j++) acc[i][j] = z8;

    // prologue: stage first tile into buffer 0
    gemm_stage_async(Asm0, Bsm0, A, Bt, K, m0, n0, 0, ldrow, ldk);
    wait_async0();
    __syncthreads();

    // K is a multiple of 64; process two 32-tiles per iteration (ping-pong)
    for (int kt = 0; kt < K; kt += 64) {
        gemm_stage_async(Asm1, Bsm1, A, Bt, K, m0, n0, kt + 32, ldrow, ldk);
        gemm_compute_step(Asm0, Bsm0, wm, wn, hl, l15, acc);
        wait_async0();
        __syncthreads();

        if (kt + 64 < K)
            gemm_stage_async(Asm0, Bsm0, A, Bt, K, m0, n0, kt + 64, ldrow, ldk);
        gemm_compute_step(Asm1, Bsm1, wm, wn, hl, l15, acc);
        wait_async0();
        __syncthreads();
    }

    // epilogue
#pragma unroll
    for (int i = 0; i < 4; i++) {
#pragma unroll
        for (int j = 0; j < 2; j++) {
            int col = n0 + wn * 32 + j * 16 + l15;
            float bv = bias[col];
#pragma unroll
            for (int e = 0; e < 8; e++) {
                int row = m0 + wm * 64 + i * 16 + e + 8 * hl;
                float v = acc[i][j][e] + bv;
                if (EPI == 1) {
                    v = gelu_exact(v);
                    ((bf16_t*)Cout)[(size_t)row * N + col] = (bf16_t)v;
                } else if (EPI == 2) {
                    v += resid[(size_t)row * N + col];
                    ((float*)Cout)[(size_t)row * N + col] = v;
                } else if (EPI == 3) {
                    // V scatter: row = b*512+s, col = h*64+d -> Vt[b][h][d][s]
                    int b = row >> 9, s = row & 511;
                    int h = col >> 6, d = col & 63;
                    ((bf16_t*)Cout)[(size_t)(((b * 16 + h) * 64 + d)) * 512 + s] = (bf16_t)v;
                } else {
                    ((bf16_t*)Cout)[(size_t)row * N + col] = (bf16_t)v;
                }
            }
        }
    }
}

// ---------------------------------------------------------------------------
// Flash attention: grid (S/64, B*H), 128 threads = 4 waves, each wave 16 q-rows
// Q[M][E] bf16, K[M][E] bf16, Vt[B,H,D,S] bf16 -> O[M][E] bf16
// ---------------------------------------------------------------------------
__global__ __launch_bounds__(128)
void attn_kernel(const bf16_t* __restrict__ Qg, const bf16_t* __restrict__ Kg,
                 const bf16_t* __restrict__ Vt, bf16_t* __restrict__ Og) {
    const int S = 512, D = 64, E = 1024, H = 16;
    __shared__ bf16_t Qs[64 * 72];
    __shared__ bf16_t KVs[128 * 72];   // K tile [key][d] (stride 72); reused as Vt tile [d][key] (stride 136)
    __shared__ bf16_t Ps[4 * 16 * 136];

    const int tid  = threadIdx.x;
    const int lane = tid & 31;
    const int wave = tid >> 5;
    const int hl   = lane >> 4;
    const int l15  = lane & 15;

    const int bh = blockIdx.y;
    const int b = bh >> 4, h = bh & 15;
    const int q0 = blockIdx.x * 64;

    // load Q tile 64x64
    {
        int r = tid >> 1, c = (tid & 1) * 32;
        const bf16_t* src = Qg + (size_t)(b * S + q0 + r) * E + h * D + c;
        bf16_t* dst = &Qs[r * 72 + c];
#pragma unroll
        for (int cc = 0; cc < 4; cc++)
            *(v8bf*)(dst + cc * 8) = *(const v8bf*)(src + cc * 8);
    }

    float m_r[8], l_r[8];
    v8f o_acc[4];
    const v8f z8 = {0.f,0.f,0.f,0.f,0.f,0.f,0.f,0.f};
#pragma unroll
    for (int e = 0; e < 8; e++) { m_r[e] = -1e30f; l_r[e] = 0.f; }
#pragma unroll
    for (int t = 0; t < 4; t++) o_acc[t] = z8;

    for (int kb = 0; kb < 4; kb++) {
        // load K tile 128x64 (stride 72)
        {
            const bf16_t* src = Kg + (size_t)(b * S + kb * 128 + tid) * E + h * D;
            bf16_t* dst = &KVs[tid * 72];
#pragma unroll
            for (int cc = 0; cc < 8; cc++)
                *(v8bf*)(dst + cc * 8) = *(const v8bf*)(src + cc * 8);
        }
        __syncthreads();

        // S = Q @ K^T for this wave's 16 rows x 128 keys
        v8f s_acc[8];
#pragma unroll
        for (int j = 0; j < 8; j++) s_acc[j] = z8;
#pragma unroll
        for (int c = 0; c < 2; c++) {
            Frag a;
            int qr = wave * 16 + l15;
            a.h[0] = *(const v8bf*)&Qs[qr * 72 + c * 32 + hl * 8];
            a.h[1] = *(const v8bf*)&Qs[qr * 72 + c * 32 + 16 + hl * 8];
#pragma unroll
            for (int j = 0; j < 8; j++) {
                Frag bb;
                int kr = j * 16 + l15;
                bb.h[0] = *(const v8bf*)&KVs[kr * 72 + c * 32 + hl * 8];
                bb.h[1] = *(const v8bf*)&KVs[kr * 72 + c * 32 + 16 + hl * 8];
                s_acc[j] = wmma_bf16(a, bb, s_acc[j]);
            }
        }

        // streaming softmax update (rows = e + 8*hl within wave stripe)
#pragma unroll
        for (int e = 0; e < 8; e++) {
            float mx = -1e30f;
#pragma unroll
            for (int j = 0; j < 8; j++) { s_acc[j][e] *= 0.125f; mx = fmaxf(mx, s_acc[j][e]); }
#pragma unroll
            for (int off = 8; off >= 1; off >>= 1) mx = fmaxf(mx, __shfl_xor(mx, off, 32));
            float mn = fmaxf(m_r[e], mx);
            float corr = __expf(m_r[e] - mn);
            l_r[e] *= corr;
#pragma unroll
            for (int t = 0; t < 4; t++) o_acc[t][e] *= corr;
            m_r[e] = mn;
            float rs = 0.f;
#pragma unroll
            for (int j = 0; j < 8; j++) {
                float p = __expf(s_acc[j][e] - mn);
                s_acc[j][e] = p; rs += p;
            }
#pragma unroll
            for (int off = 8; off >= 1; off >>= 1) rs += __shfl_xor(rs, off, 32);
            l_r[e] += rs;
            int pr = wave * 16 + e + 8 * hl;
#pragma unroll
            for (int j = 0; j < 8; j++) Ps[pr * 136 + j * 16 + l15] = (bf16_t)s_acc[j][e];
        }
        __syncthreads();   // all waves done reading K tile

        // load Vt tile 64 x 128 (stride 136) into KVs
        {
            int d = tid >> 1, c = (tid & 1) * 64;
            const bf16_t* src = Vt + (size_t)((b * H + h) * D + d) * S + kb * 128 + c;
            bf16_t* dst = &KVs[d * 136 + c];
#pragma unroll
            for (int cc = 0; cc < 8; cc++)
                *(v8bf*)(dst + cc * 8) = *(const v8bf*)(src + cc * 8);
        }
        __syncthreads();

        // O += P @ V
#pragma unroll
        for (int c2 = 0; c2 < 4; c2++) {
            Frag a;
            int pr = wave * 16 + l15;
            a.h[0] = *(const v8bf*)&Ps[pr * 136 + c2 * 32 + hl * 8];
            a.h[1] = *(const v8bf*)&Ps[pr * 136 + c2 * 32 + 16 + hl * 8];
#pragma unroll
            for (int t = 0; t < 4; t++) {
                Frag bb;
                int dr = t * 16 + l15;
                bb.h[0] = *(const v8bf*)&KVs[dr * 136 + c2 * 32 + hl * 8];
                bb.h[1] = *(const v8bf*)&KVs[dr * 136 + c2 * 32 + 16 + hl * 8];
                o_acc[t] = wmma_bf16(a, bb, o_acc[t]);
            }
        }
        __syncthreads();   // before next kb overwrites KVs
    }

#pragma unroll
    for (int t = 0; t < 4; t++) {
#pragma unroll
        for (int e = 0; e < 8; e++) {
            int row = b * S + q0 + wave * 16 + e + 8 * hl;
            int col = h * D + t * 16 + l15;
            Og[(size_t)row * E + col] = (bf16_t)(o_acc[t][e] / l_r[e]);
        }
    }
}

// ---------------------------------------------------------------------------
extern "C" void kernel_launch(void* const* d_in, const int* in_sizes, int n_in,
                              void* d_out, int out_size, void* d_ws, size_t ws_size,
                              hipStream_t stream) {
    (void)in_sizes; (void)n_in; (void)out_size; (void)ws_size;
    const int B = 16, S = 512, E = 1024, F = 4096, H = 16;
    const int M = B * S;

    const float* input = (const float*)d_in[0];
    const float* ln1_g = (const float*)d_in[1];
    const float* ln1_b = (const float*)d_in[2];
    const float* wq    = (const float*)d_in[3];
    const float* bq    = (const float*)d_in[4];
    const float* wk    = (const float*)d_in[5];
    const float* bk    = (const float*)d_in[6];
    const float* wvw   = (const float*)d_in[7];
    const float* bvb   = (const float*)d_in[8];
    const float* wo    = (const float*)d_in[9];
    const float* bo    = (const float*)d_in[10];
    const float* ln2_g = (const float*)d_in[11];
    const float* ln2_b = (const float*)d_in[12];
    const float* w1    = (const float*)d_in[13];
    const float* b1    = (const float*)d_in[14];
    const float* w2    = (const float*)d_in[15];
    const float* b2    = (const float*)d_in[16];

    char* ws = (char*)d_ws;
    const size_t MB = 1024ull * 1024ull;
    bf16_t* wq_t = (bf16_t*)(ws + 0 * MB);
    bf16_t* wk_t = (bf16_t*)(ws + 2 * MB);
    bf16_t* wv_t = (bf16_t*)(ws + 4 * MB);
    bf16_t* wo_t = (bf16_t*)(ws + 6 * MB);
    bf16_t* w1_t = (bf16_t*)(ws + 8 * MB);
    bf16_t* w2_t = (bf16_t*)(ws + 16 * MB);
    float*  xres = (float*) (ws + 24 * MB);   // residual stream x (f32, 32MB)
    bf16_t* xln  = (bf16_t*)(ws + 56 * MB);   // LN output (bf16, 16MB)
    bf16_t* qb   = (bf16_t*)(ws + 72 * MB);
    bf16_t* kbuf = (bf16_t*)(ws + 88 * MB);
    bf16_t* vtb  = (bf16_t*)(ws + 104 * MB);
    bf16_t* attn = (bf16_t*)(ws + 120 * MB);
    bf16_t* h1   = (bf16_t*)(ws + 72 * MB);   // 64MB, overlaps q/k/vt/attn (all dead by then)

    dim3 tb(32, 8);
    transpose_to_bf16<<<dim3(E / 32, E / 32), tb, 0, stream>>>(wq,  wq_t, E, E);
    transpose_to_bf16<<<dim3(E / 32, E / 32), tb, 0, stream>>>(wk,  wk_t, E, E);
    transpose_to_bf16<<<dim3(E / 32, E / 32), tb, 0, stream>>>(wvw, wv_t, E, E);
    transpose_to_bf16<<<dim3(E / 32, E / 32), tb, 0, stream>>>(wo,  wo_t, E, E);
    transpose_to_bf16<<<dim3(F / 32, E / 32), tb, 0, stream>>>(w1,  w1_t, E, F);
    transpose_to_bf16<<<dim3(E / 32, F / 32), tb, 0, stream>>>(w2,  w2_t, F, E);

    layernorm_bf16_kernel<<<M, 256, 0, stream>>>(input, ln1_g, ln1_b, xln);

    dim3 gE(E / 128, M / 128);
    gemm_bf16_kernel<0><<<gE, 256, 0, stream>>>(xln, wq_t, bq, nullptr, qb,   M, E, E);
    gemm_bf16_kernel<0><<<gE, 256, 0, stream>>>(xln, wk_t, bk, nullptr, kbuf, M, E, E);
    gemm_bf16_kernel<3><<<gE, 256, 0, stream>>>(xln, wv_t, bvb, nullptr, vtb, M, E, E);

    attn_kernel<<<dim3(S / 64, B * H), 128, 0, stream>>>(qb, kbuf, vtb, attn);

    gemm_bf16_kernel<2><<<gE, 256, 0, stream>>>(attn, wo_t, bo, input, xres, M, E, E);

    layernorm_bf16_kernel<<<M, 256, 0, stream>>>(xres, ln2_g, ln2_b, xln);

    gemm_bf16_kernel<1><<<dim3(F / 128, M / 128), 256, 0, stream>>>(xln, w1_t, b1, nullptr, h1, M, F, E);
    gemm_bf16_kernel<2><<<dim3(E / 128, M / 128), 256, 0, stream>>>(h1, w2_t, b2, xres, (float*)d_out, M, E, F);
}
